// MultiHeadAttention_64037962383811
// MI455X (gfx1250) — matrix operand order repaired and checked by reading, not compiled, not run on hardware
//
#include <hip/hip_runtime.h>
#include <hip/hip_bf16.h>
#include <stdint.h>

typedef __attribute__((ext_vector_type(16))) _Float16 v16h;
typedef __attribute__((ext_vector_type(8)))  _Float16 v8h;
typedef __attribute__((ext_vector_type(8)))  float    v8f;

#define B_  8
#define LQ_ 1024
#define LK_ 1024
#define D_  512
#define H_  8
#define HD_ (H_ * D_)   // 4096

// Assemble a 16x32 f16 A-fragment from two contiguous 8-half chunks.
// Lane (0-15|16-31) holds row M=lane%16; elements 0..7 are
// K = hlf*16 + 0..7, elements 8..15 are K = hlf*16 + 8..15 (contiguous order, same as B).
__device__ __forceinline__ v16h make_afrag(v8h lo, v8h hi) {
  union { v16h v; v8h h[2]; } u;
  u.h[0] = lo; u.h[1] = hi;
  return u.v;
}

__device__ __forceinline__ v8f wmma16(v16h a, v16h b, v8f c) {
  return __builtin_amdgcn_wmma_f32_16x16x32_f16(false, a, false, b,
                                                (short)0, c, false, false);
}

// Async copy of one 16-byte chunk from global memory directly into LDS
// (GLOBAL_LOAD_ASYNC_TO_LDS_B128, tracked by ASYNCcnt — no VGPR round trip).
__device__ __forceinline__ void async_copy16(const void* lds_dst,
                                             const void* gsrc) {
  uint32_t lds = (uint32_t)(uintptr_t)lds_dst;              // low 32b = LDS offset
  unsigned long long g = (unsigned long long)(uintptr_t)gsrc;
  asm volatile("global_load_async_to_lds_b128 %0, %1, off"
               :: "v"(lds), "v"(g) : "memory");
}

__device__ __forceinline__ void wait_async0() {
  asm volatile("s_wait_asynccnt 0" ::: "memory");
}

// ---------------------------------------------------------------------------
// Kernel 1: fp32 -> f16 conversion (grid-stride)
// ---------------------------------------------------------------------------
__global__ void f32_to_f16_kernel(const float* __restrict__ src,
                                  _Float16* __restrict__ dst, int n) {
  int i = blockIdx.x * blockDim.x + threadIdx.x;
  int stride = gridDim.x * blockDim.x;
  for (; i < n; i += stride) dst[i] = (_Float16)src[i];
}

// ---------------------------------------------------------------------------
// Kernel 2: QKV projections.
//   grid = (256 m-tiles of 32 rows over B*LQ, H, 3={Q,K,V}), block = 256.
//   Each wave computes a 32x64 strip (2 A-frags x 4 B-tiles, 8 accumulators)
//   so every B fragment is reused twice. A tile staged via async-to-LDS.
//   Q,K stored [B,H,L,D] f16; V stored transposed Vt[B,H,D,LK] f16.
// ---------------------------------------------------------------------------
__global__ __launch_bounds__(256) void qkv_proj_kernel(
    const _Float16* __restrict__ x16, const _Float16* __restrict__ s16,
    const _Float16* __restrict__ Wq16, const _Float16* __restrict__ Wk16,
    const _Float16* __restrict__ Wv16,
    const float* __restrict__ bq, const float* __restrict__ bk,
    const float* __restrict__ bv,
    _Float16* __restrict__ Qo, _Float16* __restrict__ Ko,
    _Float16* __restrict__ Vto)
{
  __shared__ _Float16 At[32][528];

  const int which = blockIdx.z;
  const int h     = blockIdx.y;
  const int m0    = blockIdx.x * 32;     // row over B*LQ (32 | 1024 -> no straddle)
  const int b     = m0 >> 10;
  const int ml    = m0 & 1023;

  const _Float16* A = (which == 0) ? x16 : s16;
  const _Float16* W = (which == 0) ? Wq16 : (which == 1 ? Wk16 : Wv16);
  const float* bias = (which == 0) ? bq : (which == 1 ? bk : bv);
  W    += (size_t)h * D_ * D_;
  bias += h * D_;

  for (int i = threadIdx.x; i < 32 * 64; i += 256) {
    int r = i >> 6, c = (i & 63) * 8;
    async_copy16(&At[r][c], A + (size_t)(m0 + r) * D_ + c);
  }
  wait_async0();
  __syncthreads();

  const int lane = threadIdx.x & 31;
  const int wave = threadIdx.x >> 5;
  const int nl   = lane & 15;
  const int hlf  = lane >> 4;
  const int n0   = wave * 64;

  v8f acc[2][4];
  { v8f z = {};
#pragma unroll
    for (int mi = 0; mi < 2; ++mi)
#pragma unroll
      for (int t = 0; t < 4; ++t) acc[mi][t] = z; }

  for (int k0 = 0; k0 < D_; k0 += 32) {
    v16h a0 = make_afrag(*(const v8h*)&At[nl][k0 + hlf * 16],
                         *(const v8h*)&At[nl][k0 + hlf * 16 + 8]);
    v16h a1 = make_afrag(*(const v8h*)&At[16 + nl][k0 + hlf * 16],
                         *(const v8h*)&At[16 + nl][k0 + hlf * 16 + 8]);
#pragma unroll
    for (int t = 0; t < 4; ++t) {
      int n = n0 + t * 16 + nl;
      v16h bf = *(const v16h*)(W + (size_t)n * D_ + k0 + hlf * 16);
      acc[0][t] = wmma16(a0, bf, acc[0][t]);
      acc[1][t] = wmma16(a1, bf, acc[1][t]);
    }
  }

  if (which < 2) {
    _Float16* base = ((which == 0) ? Qo : Ko) +
                     (((size_t)b * H_ + h) * LQ_ + ml) * D_;
#pragma unroll
    for (int mi = 0; mi < 2; ++mi)
#pragma unroll
      for (int t = 0; t < 4; ++t) {
        int n = n0 + t * 16 + nl;
        float bval = bias[n];
#pragma unroll
        for (int j = 0; j < 8; ++j) {
          int M = mi * 16 + j + hlf * 8;
          base[(size_t)M * D_ + n] = (_Float16)(acc[mi][t][j] + bval);
        }
      }
  } else {
    _Float16* base = Vto + ((size_t)b * H_ + h) * D_ * (size_t)LK_;
#pragma unroll
    for (int mi = 0; mi < 2; ++mi)
#pragma unroll
      for (int t = 0; t < 4; ++t) {
        int n = n0 + t * 16 + nl;
        float bval = bias[n];
        v8h pk;
#pragma unroll
        for (int j = 0; j < 8; ++j) pk[j] = (_Float16)(acc[mi][t][j] + bval);
        *(v8h*)(base + (size_t)n * LK_ + ml + mi * 16 + hlf * 8) = pk;
      }
  }
}

// ---------------------------------------------------------------------------
// Kernel 3: attention with online softmax.
//   grid = (LQ/16, H, B), block = 256 (8 waves). Wave w owns d-slice [64w,64w+64).
// ---------------------------------------------------------------------------
__global__ __launch_bounds__(256) void attn_kernel(
    const _Float16* __restrict__ Q, const _Float16* __restrict__ K,
    const _Float16* __restrict__ Vt, const int* __restrict__ mask,
    _Float16* __restrict__ ctx)
{
  __shared__ _Float16 Qt[16][528];
  __shared__ float    Sred[8][16][32];
  __shared__ float    Sf[16][32];
  __shared__ _Float16 P16[16][32];
  __shared__ float    mrun[16], lrun[16], alpha[16], mnsafe[16];

  const int q0 = blockIdx.x * 16;
  const int h  = blockIdx.y;
  const int b  = blockIdx.z;

  const _Float16* Qg  = Q  + (((size_t)b * H_ + h) * LQ_ + q0) * D_;
  const _Float16* Kg  = K  + ((size_t)b * H_ + h) * (size_t)LK_ * D_;
  const _Float16* Vtg = Vt + ((size_t)b * H_ + h) * (size_t)D_ * LK_;
  const int* Mg = mask + ((size_t)b * LQ_ + q0) * (size_t)LK_;

  for (int i = threadIdx.x; i < 16 * 64; i += 256) {
    int r = i >> 6, c = (i & 63) * 8;
    async_copy16(&Qt[r][c], Qg + (size_t)r * D_ + c);
  }
  if (threadIdx.x < 16) { mrun[threadIdx.x] = -__builtin_inff(); lrun[threadIdx.x] = 0.f; }
  wait_async0();
  __syncthreads();

  const int lane = threadIdx.x & 31;
  const int wave = threadIdx.x >> 5;
  const int nl   = lane & 15;
  const int hlf  = lane >> 4;
  const int d0w  = wave * 64;

  v8f acc[4];
  { v8f z = {}; acc[0] = z; acc[1] = z; acc[2] = z; acc[3] = z; }

  const float scale = 0.044194173824159216f;  // 1/sqrt(512)
  const float NEG_INF = -__builtin_inff();

  for (int kk0 = 0; kk0 < LK_; kk0 += 32) {
    // prefetch next mask block (one line per row)
    if (threadIdx.x < 16 && kk0 + 32 < LK_)
      __builtin_prefetch(&Mg[(size_t)threadIdx.x * LK_ + kk0 + 32], 0, 0);

    // ---- partial S over this wave's 64-wide d slice
    v8f s0, s1;
    { v8f z = {}; s0 = z; s1 = z; }
#pragma unroll
    for (int ks = 0; ks < 2; ++ks) {
      int d0 = d0w + ks * 32;
      v16h a = make_afrag(*(const v8h*)&Qt[nl][d0 + hlf * 16],
                          *(const v8h*)&Qt[nl][d0 + hlf * 16 + 8]);
      v16h b0 = *(const v16h*)(Kg + (size_t)(kk0 + nl) * D_ + d0 + hlf * 16);
      s0 = wmma16(a, b0, s0);
      v16h b1 = *(const v16h*)(Kg + (size_t)(kk0 + 16 + nl) * D_ + d0 + hlf * 16);
      s1 = wmma16(a, b1, s1);
    }
#pragma unroll
    for (int j = 0; j < 8; ++j) {
      Sred[wave][j + hlf * 8][nl]      = s0[j];
      Sred[wave][j + hlf * 8][16 + nl] = s1[j];
    }
    __syncthreads();  // (a)

    // ---- reduce 8 partials, scale, apply mask
    for (int slot = threadIdx.x; slot < 512; slot += 256) {
      int r = slot >> 5, c = slot & 31;
      float v = 0.f;
#pragma unroll
      for (int wv = 0; wv < 8; ++wv) v += Sred[wv][r][c];
      v *= scale;
      if (Mg[(size_t)r * LK_ + kk0 + c] == 0) v = NEG_INF;
      Sf[r][c] = v;
    }
    __syncthreads();  // (b)

    // ---- running-max update (one thread per row)
    if (threadIdx.x < 16) {
      int r = threadIdx.x;
      float mx = NEG_INF;
      for (int c = 0; c < 32; ++c) mx = fmaxf(mx, Sf[r][c]);
      float mold = mrun[r];
      float mnew = fmaxf(mold, mx);
      float msafe = (mnew == NEG_INF) ? 0.f : mnew;  // fully-masked guard
      mrun[r]   = mnew;
      mnsafe[r] = msafe;
      alpha[r]  = __expf(mold - msafe);  // -inf -> 0
    }
    __syncthreads();  // (c)

    // ---- exponentials -> P (f16 in A-fragment-friendly layout)
    for (int slot = threadIdx.x; slot < 512; slot += 256) {
      int r = slot >> 5, c = slot & 31;
      float p = __expf(Sf[r][c] - mnsafe[r]);  // -inf -> 0
      Sf[r][c]  = p;
      P16[r][c] = (_Float16)p;
    }
    __syncthreads();  // (d)

    // ---- l update (one thread per row, overlaps PV of other waves)
    if (threadIdx.x < 16) {
      int r = threadIdx.x;
      float sum = 0.f;
      for (int c = 0; c < 32; ++c) sum += Sf[r][c];
      lrun[r] = alpha[r] * lrun[r] + sum;
    }

    // ---- rescale ctx accumulators, then PV WMMAs on this wave's d slice
    float alr[8];
#pragma unroll
    for (int j = 0; j < 8; ++j) alr[j] = alpha[j + hlf * 8];
#pragma unroll
    for (int t = 0; t < 4; ++t)
#pragma unroll
      for (int j = 0; j < 8; ++j) acc[t][j] *= alr[j];

    v16h pa = make_afrag(*(const v8h*)&P16[nl][hlf * 16],
                         *(const v8h*)&P16[nl][hlf * 16 + 8]);
#pragma unroll
    for (int t = 0; t < 4; ++t) {
      int nd = d0w + t * 16 + nl;
      v16h bf = *(const v16h*)(Vtg + (size_t)nd * LK_ + kk0 + hlf * 16);
      acc[t] = wmma16(pa, bf, acc[t]);
    }
    // next iteration's first shared write (Sred) doesn't alias P16/alpha/Sf
    // reads before barrier (a), so no trailing barrier needed here.
  }
  __syncthreads();  // ensure final lrun is visible

  float lr[8];
#pragma unroll
  for (int j = 0; j < 8; ++j) {
    float l = lrun[j + hlf * 8];
    lr[j] = (l > 0.f) ? (1.f / l) : 0.f;
  }
  _Float16* cbase = ctx + ((size_t)b * LQ_ + q0) * HD_ + (size_t)h * D_;
#pragma unroll
  for (int t = 0; t < 4; ++t) {
    int nd = d0w + t * 16 + nl;
#pragma unroll
    for (int j = 0; j < 8; ++j) {
      int M = j + hlf * 8;
      cbase[(size_t)M * HD_ + nd] = (_Float16)(acc[t][j] * lr[j]);
    }
  }
}

// ---------------------------------------------------------------------------
// Kernel 4: output projection  out[8192,512] = ctx[8192,4096] @ Wp^T + bp
//   grid = 256 m-tiles of 32 rows, block = 256 (8 waves x 64 columns).
//   K is chunked by 512 through a double-buffered LDS A tile: async-to-LDS
//   loads for chunk i+1 overlap the 128 WMMAs per wave of chunk i.
// ---------------------------------------------------------------------------
__global__ __launch_bounds__(256) void out_proj_kernel(
    const _Float16* __restrict__ ctx, const _Float16* __restrict__ Wp16,
    const float* __restrict__ bp, float* __restrict__ out)
{
  __shared__ _Float16 At[2][32][528];
  const int m0   = blockIdx.x * 32;
  const int lane = threadIdx.x & 31;
  const int wave = threadIdx.x >> 5;
  const int nl   = lane & 15;
  const int hlf  = lane >> 4;
  const int n0   = wave * 64;

  auto stage = [&](int buf, int kc) {
    for (int i = threadIdx.x; i < 32 * 64; i += 256) {
      int r = i >> 6, c = (i & 63) * 8;
      async_copy16(&At[buf][r][c], ctx + (size_t)(m0 + r) * HD_ + kc + c);
    }
  };

  v8f acc[2][4];
  { v8f z = {};
#pragma unroll
    for (int mi = 0; mi < 2; ++mi)
#pragma unroll
      for (int t = 0; t < 4; ++t) acc[mi][t] = z; }

  stage(0, 0);
  wait_async0();
  __syncthreads();

  for (int kc = 0; kc < HD_; kc += 512) {
    const int buf = (kc >> 9) & 1;
    if (kc + 512 < HD_) stage(buf ^ 1, kc + 512);   // overlap with compute

    for (int k0 = 0; k0 < 512; k0 += 32) {
      v16h a0 = make_afrag(*(const v8h*)&At[buf][nl][k0 + hlf * 16],
                           *(const v8h*)&At[buf][nl][k0 + hlf * 16 + 8]);
      v16h a1 = make_afrag(*(const v8h*)&At[buf][16 + nl][k0 + hlf * 16],
                           *(const v8h*)&At[buf][16 + nl][k0 + hlf * 16 + 8]);
#pragma unroll
      for (int t = 0; t < 4; ++t) {
        int n = n0 + t * 16 + nl;
        v16h bf = *(const v16h*)(Wp16 + (size_t)n * HD_ + kc + k0 + hlf * 16);
        acc[0][t] = wmma16(a0, bf, acc[0][t]);
        acc[1][t] = wmma16(a1, bf, acc[1][t]);
      }
    }
    wait_async0();        // my async loads into buf^1 are done
    __syncthreads();      // everyone finished reading buf / staging buf^1
  }

#pragma unroll
  for (int mi = 0; mi < 2; ++mi)
#pragma unroll
    for (int t = 0; t < 4; ++t) {
      int n = n0 + t * 16 + nl;
      float bv = bp[n];
#pragma unroll
      for (int j = 0; j < 8; ++j) {
        int M = mi * 16 + j + hlf * 8;
        out[(size_t)(m0 + M) * D_ + n] = acc[mi][t][j] + bv;
      }
    }
}

// ---------------------------------------------------------------------------
extern "C" void kernel_launch(void* const* d_in, const int* in_sizes, int n_in,
                              void* d_out, int out_size, void* d_ws, size_t ws_size,
                              hipStream_t stream)
{
  const float* x   = (const float*)d_in[0];
  const float* st  = (const float*)d_in[1];
  const int*   msk = (const int*)  d_in[2];
  const float* Wq  = (const float*)d_in[3];
  const float* bq  = (const float*)d_in[4];
  const float* Wk  = (const float*)d_in[5];
  const float* bk  = (const float*)d_in[6];
  const float* Wv  = (const float*)d_in[7];
  const float* bv  = (const float*)d_in[8];
  const float* Wp  = (const float*)d_in[9];
  const float* bp  = (const float*)d_in[10];
  float* out = (float*)d_out;

  const size_t N_X  = (size_t)B_ * LQ_ * D_;        // 4 Mi
  const size_t N_W  = (size_t)H_ * D_ * D_;         // 2 Mi
  const size_t N_WP = (size_t)D_ * HD_;             // 2 Mi
  const size_t N_QK = (size_t)B_ * H_ * LQ_ * D_;   // 32 Mi
  const size_t N_CT = (size_t)B_ * LQ_ * HD_;       // 32 Mi

  _Float16* w = (_Float16*)d_ws;
  _Float16* x16  = w; w += N_X;
  _Float16* s16  = w; w += N_X;
  _Float16* wq16 = w; w += N_W;
  _Float16* wk16 = w; w += N_W;
  _Float16* wv16 = w; w += N_W;
  _Float16* wp16 = w; w += N_WP;
  _Float16* Qb   = w; w += N_QK;
  _Float16* Kb   = w; w += N_QK;
  _Float16* Vtb  = w; w += N_QK;
  _Float16* ctxb = w; w += N_CT;

  f32_to_f16_kernel<<<dim3(2048), 256, 0, stream>>>(x,  x16,  (int)N_X);
  f32_to_f16_kernel<<<dim3(2048), 256, 0, stream>>>(st, s16,  (int)N_X);
  f32_to_f16_kernel<<<dim3(2048), 256, 0, stream>>>(Wq, wq16, (int)N_W);
  f32_to_f16_kernel<<<dim3(2048), 256, 0, stream>>>(Wk, wk16, (int)N_W);
  f32_to_f16_kernel<<<dim3(2048), 256, 0, stream>>>(Wv, wv16, (int)N_W);
  f32_to_f16_kernel<<<dim3(2048), 256, 0, stream>>>(Wp, wp16, (int)N_WP);

  qkv_proj_kernel<<<dim3((B_ * LQ_) / 32, H_, 3), 256, 0, stream>>>(
      x16, s16, wq16, wk16, wv16, bq, bk, bv, Qb, Kb, Vtb);

  attn_kernel<<<dim3(LQ_ / 16, H_, B_), 256, 0, stream>>>(Qb, Kb, Vtb, msk, ctxb);

  out_proj_kernel<<<dim3((B_ * LQ_) / 32), 256, 0, stream>>>(ctxb, wp16, bp, out);
}
